// Parser_39419209843231
// MI455X (gfx1250) — matrix operand order repaired and checked by reading, not compiled, not run on hardware
//
#include <hip/hip_runtime.h>

// ---------------------------------------------------------------------------
// Model dims (compile-time, match reference)
// ---------------------------------------------------------------------------
#define BB   4
#define SS   256
#define HH   512
#define GHH  2048      // 4*H
#define DINN 1024      // 2*E == 2*H  (same K for both layers)

typedef __attribute__((ext_vector_type(16))) __bf16 v16bf;
typedef __attribute__((ext_vector_type(8)))  float  v8f;

struct B256 { uint4 a, b; };

static __device__ __forceinline__ v16bf make_frag(uint4 a, uint4 b) {
  B256 t{a, b};
  return __builtin_bit_cast(v16bf, t);
}

static __device__ __forceinline__ unsigned short f2bf(float f) {
  unsigned int u = __builtin_bit_cast(unsigned int, f);
  u += 0x7FFFu + ((u >> 16) & 1u);            // round-to-nearest-even
  return (unsigned short)(u >> 16);
}

static __device__ __forceinline__ float sigf(float x) {
  return 1.0f / (1.0f + __expf(-x));
}
static __device__ __forceinline__ float tanhf_fast(float x) {
  // branch-free, inf-safe: tanh(x) = 1 - 2/(e^{2x}+1)
  return 1.0f - 2.0f / (__expf(2.0f * x) + 1.0f);
}

// ---------------------------------------------------------------------------
// fp32 -> bf16 conversion (weights), grid-stride
// ---------------------------------------------------------------------------
__global__ void __launch_bounds__(256)
cvt_bf16(const float* __restrict__ src, unsigned short* __restrict__ dst, int n) {
  for (int i = blockIdx.x * 256 + threadIdx.x; i < n; i += gridDim.x * 256)
    dst[i] = f2bf(src[i]);
}

// ---------------------------------------------------------------------------
// Embedding gather + concat -> x0 bf16, row = s*B+b, 1024 cols
// ---------------------------------------------------------------------------
__global__ void __launch_bounds__(256)
embed_concat(const int* __restrict__ sent, const int* __restrict__ pos,
             const float* __restrict__ emb, unsigned short* __restrict__ xo) {
  int idx = blockIdx.x * 256 + threadIdx.x;            // 1024*1024 total
  if (idx >= BB * SS * DINN) return;
  int row = idx >> 10;                                 // s*4+b
  int d   = idx & 1023;
  int s = row >> 2, b = row & 3;
  int tok = (d < HH) ? sent[b * SS + s] : pos[b * SS + s];
  float v = emb[(size_t)tok * HH + (d & (HH - 1))];
  xo[idx] = f2bf(v);
}

// ---------------------------------------------------------------------------
// Input-projection GEMM: pre[row][g] = sum_k A[row][k]*W[g][k] + bih[g]+bhh[g]
// A: [1024][1024] bf16, W: [2048][1024] bf16, pre: [1024][2048] f32
// One wave -> 16x64 strip. 2048 waves = 256 blocks * 8 waves.
// ---------------------------------------------------------------------------
__global__ void __launch_bounds__(256)
gemm_pre(const unsigned short* __restrict__ A,
         const unsigned short* __restrict__ W,
         const float* __restrict__ bih, const float* __restrict__ bhh,
         float* __restrict__ pre) {
  const int wave = blockIdx.x * 8 + (threadIdx.x >> 5);
  const int lane = threadIdx.x & 31;
  const int half = lane >> 4, n = lane & 15;
  const int mt    = wave & 63;           // 64 M-tiles
  const int nbase = (wave >> 6) * 64;    // 32 N-groups of 64 cols

  v8f acc[4] = {v8f{}, v8f{}, v8f{}, v8f{}};
  const unsigned short* arow = A + (size_t)(mt * 16 + n) * DINN;
  for (int k0 = 0; k0 < DINN; k0 += 32) {
    uint4 a0 = *(const uint4*)(arow + k0 + 8 * half);
    uint4 a1 = *(const uint4*)(arow + k0 + 16 + 8 * half);
    v16bf af = make_frag(a0, a1);
#pragma unroll
    for (int j = 0; j < 4; ++j) {
      const unsigned short* wr =
          W + (size_t)(nbase + j * 16 + n) * DINN + k0 + 8 * half;
      v16bf bf = make_frag(*(const uint4*)wr, *(const uint4*)(wr + 16));
      acc[j] = __builtin_amdgcn_wmma_f32_16x16x32_bf16(
          false, af, false, bf, (short)0, acc[j], false, false);
    }
  }
#pragma unroll
  for (int j = 0; j < 4; ++j) {
    int col = nbase + j * 16 + n;
    float bias = bih[col] + bhh[col];
#pragma unroll
    for (int r = 0; r < 8; ++r) {
      int row = mt * 16 + r + 8 * half;
      pre[(size_t)row * GHH + col] = acc[j][r] + bias;
    }
  }
}

// ---------------------------------------------------------------------------
// Persistent recurrence: grid.x=2 (dir), 512 threads = 16 waves.
// Per step: G[4][2048] = h @ W_hh^T (WMMA, M padded to 16), gates, masked
// state update; h (bf16) & c (f32) live in LDS across all 256 steps.
// Output written as bf16 into the concat half of the next layer's input.
// ---------------------------------------------------------------------------
__global__ void __launch_bounds__(512)
lstm_scan(const unsigned short* __restrict__ whh_f,
          const unsigned short* __restrict__ whh_b,
          const float* __restrict__ pre_f, const float* __restrict__ pre_b,
          const int* __restrict__ length, unsigned short* __restrict__ xout) {
  __shared__ __align__(16) unsigned short hbuf[16 * HH];  // rows 4..15 stay 0
  __shared__ float cbuf[BB * HH];
  __shared__ float gbuf[BB * GHH];

  const int dir = blockIdx.x;
  const unsigned short* whh = dir ? whh_b : whh_f;
  const float* pre = dir ? pre_b : pre_f;
  const int tid = threadIdx.x;

  for (int i = tid; i < 16 * HH; i += 512) hbuf[i] = 0;
  for (int i = tid; i < BB * HH; i += 512) cbuf[i] = 0.0f;
  int len0 = length[0], len1 = length[1], len2 = length[2], len3 = length[3];
  __syncthreads();

  const int wave = tid >> 5;
  const int lane = tid & 31;
  const int half = lane >> 4, n = lane & 15;

  for (int t = 0; t < SS; ++t) {
    const int s = dir ? (SS - 1 - t) : t;

    // ---- phase A: G = h @ W_hh^T (+pre), each wave owns 128 gate columns
#pragma unroll 1
    for (int tile = 0; tile < 8; ++tile) {
      const int col0 = wave * 128 + tile * 16;
      v8f acc = {};
      const unsigned short* hrow = &hbuf[n * HH];
      const unsigned short* wrow = whh + (size_t)(col0 + n) * HH + 8 * half;
#pragma unroll 4
      for (int k0 = 0; k0 < HH; k0 += 32) {
        uint4 a0 = *(const uint4*)(hrow + k0 + 8 * half);
        uint4 a1 = *(const uint4*)(hrow + k0 + 16 + 8 * half);
        v16bf af = make_frag(a0, a1);
        v16bf bf = make_frag(*(const uint4*)(wrow + k0),
                             *(const uint4*)(wrow + k0 + 16));
        acc = __builtin_amdgcn_wmma_f32_16x16x32_bf16(
            false, af, false, bf, (short)0, acc, false, false);
      }
      if (half == 0) {            // only M rows 0..3 are real batch rows
        const int col = col0 + n;
#pragma unroll
        for (int r = 0; r < BB; ++r)
          gbuf[r * GHH + col] = acc[r] + pre[(size_t)(s * BB + r) * GHH + col];
      }
    }
    __syncthreads();

    // ---- phase B: gates -> c,h update (masked), emit output
    {
      const int j = tid;  // hidden unit
#pragma unroll
      for (int b = 0; b < BB; ++b) {
        float gi = gbuf[b * GHH + j];
        float gf = gbuf[b * GHH + HH + j];
        float gg = gbuf[b * GHH + 2 * HH + j];
        float go = gbuf[b * GHH + 3 * HH + j];
        float cold = cbuf[b * HH + j];
        float cn = sigf(gf) * cold + sigf(gi) * tanhf_fast(gg);
        float hn = sigf(go) * tanhf_fast(cn);
        int lb = (b == 0) ? len0 : (b == 1) ? len1 : (b == 2) ? len2 : len3;
        bool m = s < lb;
        if (m) { cbuf[b * HH + j] = cn; hbuf[b * HH + j] = f2bf(hn); }
        float outv = m ? hn : 0.0f;
        xout[(size_t)(s * BB + b) * DINN + dir * HH + j] = f2bf(outv);
      }
    }
    __syncthreads();
  }
}

// ---------------------------------------------------------------------------
// pa/pb projection: out[row][h] = sum_k X[row][k] * w1[h][colOff + k]
// X: [1024][1024] bf16, w1b: [512][2048] bf16. blocks 0..63 -> pa, 64..127 -> pb
// ---------------------------------------------------------------------------
__global__ void __launch_bounds__(256)
gemm_pab(const unsigned short* __restrict__ X,
         const unsigned short* __restrict__ w1b,
         float* __restrict__ pa, float* __restrict__ pb) {
  int blk = blockIdx.x;
  const int which = (blk >= 64);
  blk -= which * 64;
  float* out = which ? pb : pa;
  const int colOff = which * DINN;

  const int wave = blk * 8 + (threadIdx.x >> 5);   // 512 waves
  const int lane = threadIdx.x & 31;
  const int half = lane >> 4, n = lane & 15;
  const int mt    = wave & 63;
  const int nbase = (wave >> 6) * 64;              // 8 N-groups over 512

  v8f acc[4] = {v8f{}, v8f{}, v8f{}, v8f{}};
  const unsigned short* arow = X + (size_t)(mt * 16 + n) * DINN;
  for (int k0 = 0; k0 < DINN; k0 += 32) {
    uint4 a0 = *(const uint4*)(arow + k0 + 8 * half);
    uint4 a1 = *(const uint4*)(arow + k0 + 16 + 8 * half);
    v16bf af = make_frag(a0, a1);
#pragma unroll
    for (int j = 0; j < 4; ++j) {
      const unsigned short* wr =
          w1b + (size_t)(nbase + j * 16 + n) * GHH + colOff + k0 + 8 * half;
      v16bf bf = make_frag(*(const uint4*)wr, *(const uint4*)(wr + 16));
      acc[j] = __builtin_amdgcn_wmma_f32_16x16x32_bf16(
          false, af, false, bf, (short)0, acc[j], false, false);
    }
  }
#pragma unroll
  for (int j = 0; j < 4; ++j) {
    int col = nbase + j * 16 + n;
#pragma unroll
    for (int r = 0; r < 8; ++r) {
      int row = mt * 16 + r + 8 * half;
      out[(size_t)row * HH + col] = acc[j][r];
    }
  }
}

// ---------------------------------------------------------------------------
// score[b,s,t] = sum_h tanh(pa[b,s,h] + pb[b,t,h] + b1[h]) * w2[h] + b2
// one block per (b,s); thread t owns one output column
// ---------------------------------------------------------------------------
__global__ void __launch_bounds__(256)
attn_score(const float* __restrict__ pa, const float* __restrict__ pb,
           const float* __restrict__ b1, const float* __restrict__ w2,
           const float* __restrict__ b2, float* __restrict__ out) {
  __shared__ float ppa[HH];
  __shared__ float w2s[HH];
  const int blk = blockIdx.x;
  const int b = blk >> 8, s = blk & 255;
  const int tid = threadIdx.x;
  for (int h = tid; h < HH; h += 256) {
    ppa[h] = pa[(size_t)(s * BB + b) * HH + h] + b1[h];
    w2s[h] = w2[h];
  }
  __syncthreads();
  const int t = tid;
  const float* pbr = pb + (size_t)(t * BB + b) * HH;
  float acc = 0.0f;
#pragma unroll 8
  for (int h = 0; h < HH; ++h) {
    float v = ppa[h] + pbr[h];
    acc = __builtin_fmaf(tanhf_fast(v), w2s[h], acc);
  }
  out[((size_t)(b * SS + s)) * SS + t] = acc + b2[0];
}

// ---------------------------------------------------------------------------
// Host-side orchestration
// ---------------------------------------------------------------------------
extern "C" void kernel_launch(void* const* d_in, const int* in_sizes, int n_in,
                              void* d_out, int out_size, void* d_ws, size_t ws_size,
                              hipStream_t stream) {
  (void)in_sizes; (void)n_in; (void)out_size; (void)ws_size;
  const int*   sent   = (const int*)d_in[0];
  const int*   pos    = (const int*)d_in[1];
  const int*   length = (const int*)d_in[2];
  const float* emb    = (const float*)d_in[3];
  const float* w_ih   = (const float*)d_in[4];   // [2][2][2048][1024]
  const float* w_hh   = (const float*)d_in[5];   // [2][2][2048][512]
  const float* b_ih   = (const float*)d_in[6];   // [2][2][2048]
  const float* b_hh   = (const float*)d_in[7];
  const float* w1     = (const float*)d_in[8];   // [512][2048]
  const float* b1     = (const float*)d_in[9];
  const float* w2     = (const float*)d_in[10];  // [1][512]
  const float* b2     = (const float*)d_in[11];
  float* out = (float*)d_out;

  // workspace carve-out
  char* ws = (char*)d_ws;
  size_t off = 0;
  auto carve = [&](size_t bytes) -> void* {
    void* p = ws + off;
    off += (bytes + 255) & ~(size_t)255;
    return p;
  };
  const size_t N_WIH = (size_t)2 * 2 * GHH * DINN;  // 8.4M
  const size_t N_WHH = (size_t)2 * 2 * GHH * HH;    // 4.2M
  const size_t N_W1  = (size_t)HH * GHH;            // 1.0M
  unsigned short* wihb = (unsigned short*)carve(N_WIH * 2);
  unsigned short* whhb = (unsigned short*)carve(N_WHH * 2);
  unsigned short* w1b  = (unsigned short*)carve(N_W1 * 2);
  unsigned short* xa   = (unsigned short*)carve((size_t)BB * SS * DINN * 2);
  unsigned short* xb   = (unsigned short*)carve((size_t)BB * SS * DINN * 2);
  float* preF = (float*)carve((size_t)BB * SS * GHH * 4);
  float* preB = (float*)carve((size_t)BB * SS * GHH * 4);
  float* pa   = (float*)carve((size_t)BB * SS * HH * 4);
  float* pb   = (float*)carve((size_t)BB * SS * HH * 4);

  // 1) weight conversion to bf16
  cvt_bf16<<<2048, 256, 0, stream>>>(w_ih, wihb, (int)N_WIH);
  cvt_bf16<<<2048, 256, 0, stream>>>(w_hh, whhb, (int)N_WHH);
  cvt_bf16<<<1024, 256, 0, stream>>>(w1,   w1b,  (int)N_W1);

  // 2) embedding gather + concat
  embed_concat<<<(BB * SS * DINN) / 256, 256, 0, stream>>>(sent, pos, emb, xa);

  const size_t WIH_LD = (size_t)GHH * DINN;  // per (l,dir)
  const size_t WHH_LD = (size_t)GHH * HH;
  // 3) layer 0
  gemm_pre<<<256, 256, 0, stream>>>(xa, wihb + 0 * WIH_LD, b_ih + 0 * GHH,
                                    b_hh + 0 * GHH, preF);
  gemm_pre<<<256, 256, 0, stream>>>(xa, wihb + 1 * WIH_LD, b_ih + 1 * GHH,
                                    b_hh + 1 * GHH, preB);
  lstm_scan<<<2, 512, 0, stream>>>(whhb + 0 * WHH_LD, whhb + 1 * WHH_LD,
                                   preF, preB, length, xb);
  // 4) layer 1
  gemm_pre<<<256, 256, 0, stream>>>(xb, wihb + 2 * WIH_LD, b_ih + 2 * GHH,
                                    b_hh + 2 * GHH, preF);
  gemm_pre<<<256, 256, 0, stream>>>(xb, wihb + 3 * WIH_LD, b_ih + 3 * GHH,
                                    b_hh + 3 * GHH, preB);
  lstm_scan<<<2, 512, 0, stream>>>(whhb + 2 * WHH_LD, whhb + 3 * WHH_LD,
                                   preF, preB, length, xa);
  // 5) attention projections + score
  gemm_pab<<<128, 256, 0, stream>>>(xa, w1b, pa, pb);
  attn_score<<<BB * SS, 256, 0, stream>>>(pa, pb, b1, w2, b2, out);
}
